// ArcPredictorWloss_69801808495254
// MI455X (gfx1250) — matrix operand order repaired and checked by reading, not compiled, not run on hardware
//
#include <hip/hip_runtime.h>
#include <hip/hip_bf16.h>

#define NS   16
#define L    256
#define LL   65536
#define H    128
#define HID  128
#define LAMW 20.0f
#define NIT  20
#define EPSW 1e-8f

#if __has_builtin(__builtin_amdgcn_global_load_async_to_lds_b128)
#define ASYNC_LDS 1
#else
#define ASYNC_LDS 0
#endif

typedef __bf16 bf16x16 __attribute__((ext_vector_type(16)));
typedef float f32x8 __attribute__((ext_vector_type(8)));
typedef unsigned short u16x8 __attribute__((ext_vector_type(8)));
typedef unsigned short u16x16 __attribute__((ext_vector_type(16)));
typedef int v4i __attribute__((vector_size(16)));
typedef __attribute__((address_space(1))) v4i* gbl_v4i_p;
typedef __attribute__((address_space(3))) v4i* lds_v4i_p;

__device__ __forceinline__ unsigned short f2bf(float x) {
  unsigned u = __float_as_uint(x);
  unsigned r = u + 0x7FFFu + ((u >> 16) & 1u);   // round-to-nearest-even
  return (unsigned short)(r >> 16);
}

// branchless tanh: 1 - 2/(e^{2x}+1); exact at +-inf, no exec-mask branches
__device__ __forceinline__ float tanh_fast(float x) {
  float e = __expf(2.0f * x);
  return 1.0f - 2.0f * __builtin_amdgcn_rcpf(e + 1.0f);
}

__device__ __forceinline__ f32x8 wmma_bf16(bf16x16 a, bf16x16 b, f32x8 c) {
  return __builtin_amdgcn_wmma_f32_16x16x32_bf16(false, a, false, b,
                                                 (short)0, c, false, false);
}

// 16 consecutive bf16 (B-fragment: lane = col N=lane%16, K=16*(lane>=16)+e)
__device__ __forceinline__ bf16x16 ldsB16(const unsigned short* p) {
  u16x8 lo = *(const u16x8*)p;
  u16x8 hi = *(const u16x8*)(p + 8);
  u16x16 u;
#pragma unroll
  for (int e = 0; e < 8; ++e) { u[e] = lo[e]; u[8 + e] = hi[e]; }
  return __builtin_bit_cast(bf16x16, u);
}

// A-fragment from an f32 row (global or LDS): elems 0..15 @k0..k0+15
__device__ __forceinline__ bf16x16 cvtA16(const float* row, int k0) {
  float4 x0 = *(const float4*)(row + k0);
  float4 x1 = *(const float4*)(row + k0 + 4);
  float4 x2 = *(const float4*)(row + k0 + 8);
  float4 x3 = *(const float4*)(row + k0 + 12);
  u16x16 u;
  u[0]=f2bf(x0.x); u[1]=f2bf(x0.y); u[2]=f2bf(x0.z); u[3]=f2bf(x0.w);
  u[4]=f2bf(x1.x); u[5]=f2bf(x1.y); u[6]=f2bf(x1.z); u[7]=f2bf(x1.w);
  u[8]=f2bf(x2.x); u[9]=f2bf(x2.y); u[10]=f2bf(x2.z); u[11]=f2bf(x2.w);
  u[12]=f2bf(x3.x); u[13]=f2bf(x3.y); u[14]=f2bf(x3.z); u[15]=f2bf(x3.w);
  return __builtin_bit_cast(bf16x16, u);
}

// ---------------------------------------------------------------- init
__global__ void init_kernel(unsigned int* maxabs) {
  if (threadIdx.x < NS) maxabs[threadIdx.x] = 0u;
}

// -------------------------------------------------- MLP scores + cost C
// grid: 16 samples * 512 blocks; block 256 threads (8 waves); wave = 16 arcs
__global__ __launch_bounds__(256)
void mlp_cost_kernel(const float* __restrict__ f, const float* __restrict__ W1,
                     const float* __restrict__ b1, const float* __restrict__ W2,
                     const float* __restrict__ b2, float* __restrict__ scores,
                     float* __restrict__ Cbuf, unsigned int* __restrict__ maxabs) {
  __shared__ __attribute__((aligned(16))) unsigned short w1T[HID][H + 8]; // W1 col-major bf16
  __shared__ __attribute__((aligned(16))) float tile[8][16][H + 4];       // f-tile, then hdn
  __shared__ float b1s[HID], w2s[HID];

  const int tid = threadIdx.x, wave = tid >> 5, lane = tid & 31;
  const int wg = blockIdx.x;
  const int s = wg >> 9;
  const int arc0 = (wg & 511) * 128;

  const int arcW = arc0 + wave * 16;
  const float* fbase = f + ((size_t)s * LL + arcW) * H;

  // stage 16 arc feature rows -> LDS f32 (async DMA path on CDNA5)
#if ASYNC_LDS
#pragma unroll
  for (int r = 0; r < 16; ++r) {
    const float* g = fbase + r * H + lane * 4;
    float* l = &tile[wave][r][lane * 4];
    __builtin_amdgcn_global_load_async_to_lds_b128((gbl_v4i_p)g, (lds_v4i_p)l, 0, 0);
  }
#else
#pragma unroll 4
  for (int r = 0; r < 16; ++r) {
    float4 v = *(const float4*)(fbase + r * H + lane * 4);
    *(float4*)&tile[wave][r][lane * 4] = v;
  }
#endif

  for (int idx = tid; idx < H * HID; idx += 256) {
    int k = idx >> 7, j = idx & 127;           // W1[k][j] row-major
    w1T[j][k] = f2bf(W1[idx]);
  }
  if (tid < HID) { b1s[tid] = b1[tid]; w2s[tid] = W2[tid]; }

#if ASYNC_LDS
#if __has_builtin(__builtin_amdgcn_s_wait_asynccnt)
  __builtin_amdgcn_s_wait_asynccnt(0);
#else
  asm volatile("s_wait_asynccnt 0" ::: "memory");
#endif
#endif
  __syncthreads();

  // preload 4 A fragments (K = 128) from the f32 tile
  bf16x16 afr[4];
  {
    const float* row = tile[wave][lane & 15];
    int kh = (lane & 16) ? 16 : 0;
#pragma unroll
    for (int kb = 0; kb < 4; ++kb) afr[kb] = cvtA16(row, kb * 32 + kh);
  }

  const int ncol = lane & 15;
  const int khiB = (lane & 16) ? 16 : 0;
  for (int nt = 0; nt < 8; ++nt) {
    f32x8 acc = {};
#pragma unroll
    for (int kb = 0; kb < 4; ++kb)
      acc = wmma_bf16(afr[kb], ldsB16(&w1T[nt * 16 + ncol][kb * 32 + khiB]), acc);
    float bias = b1s[nt * 16 + ncol];
#pragma unroll
    for (int v = 0; v < 8; ++v) {
      int row = v + ((lane & 16) ? 8 : 0);
      tile[wave][row][nt * 16 + ncol] = tanh_fast(acc[v] + bias);  // tile now = hdn
    }
  }

  // layer 2: score[arc] = hdn . W2 + b2
  {
    int row = lane & 15, c0 = (lane >> 4) * 64;
    float p = 0.f;
#pragma unroll 8
    for (int c = 0; c < 64; ++c) p += tile[wave][row][c0 + c] * w2s[c0 + c];
    p += __shfl_xor(p, 16, 32);
    if (lane < 16) scores[(size_t)s * LL + arcW + row] = p + b2[0];
  }

  // cost C[arc] = dot(f[n,i,j,:], f[n,j,i,:]) + per-sample max|C|
  float mloc = 0.f;
  for (int a = 0; a < 16; ++a) {
    int arc = arcW + a, i = arc >> 8, j = arc & 255;
    const float* own = f + ((size_t)s * LL + arc) * H;
    const float* par = f + ((size_t)s * LL + (j << 8) + i) * H;
    if (a + 1 < 16) {   // prefetch next transpose-partner row (strided)
      int arcn = arc + 1, jn = arcn & 255, in = arcn >> 8;
      __builtin_prefetch(f + ((size_t)s * LL + (jn << 8) + in) * H + lane * 4, 0, 0);
    }
    float4 o = *(const float4*)(own + lane * 4);
    float4 q = *(const float4*)(par + lane * 4);
    float d = o.x * q.x + o.y * q.y + o.z * q.z + o.w * q.w;
#pragma unroll
    for (int m = 1; m < 32; m <<= 1) d += __shfl_xor(d, m, 32);
    if (lane == 0) Cbuf[(size_t)s * LL + arc] = d;
    mloc = fmaxf(mloc, fabsf(d));
  }
  if (lane == 0) atomicMax(&maxabs[s], __float_as_uint(mloc));
}

// ------------------------------------------- row distributions r and c
__global__ __launch_bounds__(256)
void rowdist_kernel(const float* __restrict__ scores, const float* __restrict__ head,
                    float* __restrict__ rbuf, float* __restrict__ cbuf) {
  __shared__ float red[256];
  const int row = blockIdx.x, j = threadIdx.x;
  float sc = scores[(size_t)row * L + j];
  red[j] = sc; __syncthreads();
  for (int st = 128; st > 0; st >>= 1) {
    if (j < st) red[j] = fmaxf(red[j], red[j + st]);
    __syncthreads();
  }
  float mx = red[0]; __syncthreads();
  float e = __expf(sc - mx);
  red[j] = e; __syncthreads();
  for (int st = 128; st > 0; st >>= 1) {
    if (j < st) red[j] += red[j + st];
    __syncthreads();
  }
  float se = red[0]; __syncthreads();
  rbuf[(size_t)row * L + j] = e * __builtin_amdgcn_rcpf(se);
  float t = head[(size_t)row * L + j] + EPSW;
  red[j] = t; __syncthreads();
  for (int st = 128; st > 0; st >>= 1) {
    if (j < st) red[j] += red[j + st];
    __syncthreads();
  }
  cbuf[(size_t)row * L + j] = t * __builtin_amdgcn_rcpf(red[0]);
}

// --------------------------------------------------- batched Sinkhorn
// out[row,:] = dist[row,:] / (in[row,:] @ Kop + eps); Kop given by its
// "transpose-row-major" LDS image Kl (Kl[c][k] supplies B-fragments).
__device__ __forceinline__ void sink_pass(const float* inM,
                                          const unsigned short (*Kl)[L + 8],
                                          const float* dist, float* outM,
                                          int lane) {
  bf16x16 af[8];
  const float* rp = inM + (lane & 15) * L;
  const int kha = (lane & 16) ? 16 : 0;
#pragma unroll
  for (int kb = 0; kb < 8; ++kb) af[kb] = cvtA16(rp, kb * 32 + kha);

  const int ncol = lane & 15;
  const int khb = (lane & 16) ? 16 : 0;
  for (int nt = 0; nt < 16; ++nt) {
    f32x8 acc = {};
#pragma unroll
    for (int kb = 0; kb < 8; ++kb)
      acc = wmma_bf16(af[kb], ldsB16(&Kl[nt * 16 + ncol][kb * 32 + khb]), acc);
#pragma unroll
    for (int v = 0; v < 8; ++v) {
      int row = v + ((lane & 16) ? 8 : 0);
      int col = nt * 16 + ncol;
      outM[row * L + col] = dist[row * L + col] *
                            __builtin_amdgcn_rcpf(acc[v] + EPSW);
    }
  }
}

// grid: 16 samples * 8 row-groups; block 64 threads (2 waves, 16 rows each)
__global__ __launch_bounds__(64)
void sinkhorn_kernel(const float* __restrict__ Cbuf, const float* __restrict__ rbuf,
                     const float* __restrict__ cbuf, float* __restrict__ Ubuf,
                     float* __restrict__ Abuf, const unsigned int* __restrict__ maxabs,
                     float* __restrict__ partials) {
  __shared__ __attribute__((aligned(16))) unsigned short Ks[L][L + 8];  // K row-major (later KM)
  __shared__ __attribute__((aligned(16))) unsigned short KTs[L][L + 8]; // K col-major
  __shared__ float wpart[2];

  const int blk = blockIdx.x, s = blk >> 3, rg = blk & 7;
  const int tid = threadIdx.x, wave = tid >> 5, lane = tid & 31;
  const float invmax = 1.f / (__uint_as_float(maxabs[s]) + EPSW);
  const float* Cs = Cbuf + (size_t)s * LL;

  for (int idx = tid; idx < LL; idx += 64) {
    int i = idx >> 8, j = idx & 255;
    float cn = Cs[idx] * invmax;
    unsigned short kb = f2bf(__expf(-LAMW * cn));
    Ks[i][j] = kb;
    KTs[j][i] = kb;
  }
  __syncthreads();

  const int rowBase = rg * 32 + wave * 16;
  const float* rS = rbuf + ((size_t)s * L + rowBase) * L;
  const float* cS = cbuf + ((size_t)s * L + rowBase) * L;
  float* US = Ubuf + ((size_t)s * L + rowBase) * L;
  float* AS = Abuf + ((size_t)s * L + rowBase) * L;

  for (int idx = lane; idx < 16 * L; idx += 32) US[idx] = 1.0f / L;

  for (int it = 0; it < NIT; ++it) {
    sink_pass(US, KTs, cS, AS, lane);  // A = c / (U@K + eps)
    sink_pass(AS, Ks, rS, US, lane);   // U = r / (A@K^T + eps)
  }
  sink_pass(US, KTs, cS, AS, lane);    // V = c / (U@K + eps)  -> AS

  __syncthreads();                     // all waves done with Ks as K
  for (int idx = tid; idx < LL; idx += 64) {
    int i = idx >> 8, j = idx & 255;
    float cn = Cs[idx] * invmax;
    Ks[i][j] = f2bf(__expf(-LAMW * cn) * cn);   // KM = K * M
  }
  __syncthreads();

  // X = V @ KM^T ; partial = sum(U * X)
  float part = 0.f;
  {
    bf16x16 af[8];
    const float* rp = AS + (lane & 15) * L;
    const int kha = (lane & 16) ? 16 : 0;
#pragma unroll
    for (int kb = 0; kb < 8; ++kb) af[kb] = cvtA16(rp, kb * 32 + kha);
    const int ncol = lane & 15;
    const int khb = (lane & 16) ? 16 : 0;
    for (int nt = 0; nt < 16; ++nt) {
      f32x8 acc = {};
#pragma unroll
      for (int kb = 0; kb < 8; ++kb)
        acc = wmma_bf16(af[kb], ldsB16(&Ks[nt * 16 + ncol][kb * 32 + khb]), acc);
#pragma unroll
      for (int v = 0; v < 8; ++v) {
        int row = v + ((lane & 16) ? 8 : 0);
        part += acc[v] * US[row * L + nt * 16 + ncol];
      }
    }
  }
#pragma unroll
  for (int m = 1; m < 32; m <<= 1) part += __shfl_xor(part, m, 32);
  if (lane == 0) wpart[wave] = part;
  __syncthreads();
  if (tid == 0) partials[blk] = wpart[0] + wpart[1];
}

// --------------------------------------------------------- final reduce
__global__ __launch_bounds__(128)
void loss_reduce_kernel(const float* __restrict__ partials, float* __restrict__ loss) {
  __shared__ float sh[128];
  int t = threadIdx.x;
  sh[t] = partials[t];
  __syncthreads();
  for (int st = 64; st > 0; st >>= 1) {
    if (t < st) sh[t] += sh[t + st];
    __syncthreads();
  }
  if (t == 0) *loss = sh[0];
}

extern "C" void kernel_launch(void* const* d_in, const int* in_sizes, int n_in,
                              void* d_out, int out_size, void* d_ws, size_t ws_size,
                              hipStream_t stream) {
  const float* f    = (const float*)d_in[0];
  const float* head = (const float*)d_in[1];
  const float* W1   = (const float*)d_in[2];
  const float* b1   = (const float*)d_in[3];
  const float* W2   = (const float*)d_in[4];
  const float* b2   = (const float*)d_in[5];

  float* scores = (float*)d_out;                    // (16,256,256)
  float* loss   = scores + (size_t)NS * LL;         // scalar

  float* ws   = (float*)d_ws;
  float* Cbuf = ws;                                 // 16*65536
  float* rbuf = Cbuf + (size_t)NS * LL;
  float* cbuf = rbuf + (size_t)NS * LL;
  float* Ubuf = cbuf + (size_t)NS * LL;
  float* Abuf = Ubuf + (size_t)NS * LL;
  unsigned int* maxabs = (unsigned int*)(Abuf + (size_t)NS * LL);
  float* partials = (float*)(maxabs + 32);

  init_kernel<<<1, 32, 0, stream>>>(maxabs);
  mlp_cost_kernel<<<NS * 512, 256, 0, stream>>>(f, W1, b1, W2, b2, scores, Cbuf, maxabs);
  rowdist_kernel<<<NS * L, 256, 0, stream>>>(scores, head, rbuf, cbuf);
  sinkhorn_kernel<<<NS * 8, 64, 0, stream>>>(Cbuf, rbuf, cbuf, Ubuf, Abuf, maxabs, partials);
  loss_reduce_kernel<<<1, 128, 0, stream>>>(partials, loss);
}
